// ADMEModel_29240137351459
// MI455X (gfx1250) — matrix operand order repaired and checked by reading, not compiled, not run on hardware
//
#include <hip/hip_runtime.h>
#include <hip/hip_bf16.h>
#include <math.h>

#define NNODE  65536
#define NEDGE  262144
#define NBATCH 2048
#define EN_TOT (NEDGE + NNODE)   // edges + self loops for GAT

typedef __attribute__((ext_vector_type(16))) __bf16 v16bf;
typedef __attribute__((ext_vector_type(8)))  float  v8f;

__device__ __forceinline__ float sigmoidf(float v) {
  return __builtin_amdgcn_rcpf(1.0f + __expf(-v));   // v_rcp_f32, not div-scale sequence
}

// ---------------- WMMA bf16 A-fragment loader (contiguous K order shared with B, wave32) ----------------
// A 16x32 (MxK): lane&15 = M; lanes 0-15 hold K 0..15, lanes 16-31 hold K 16..31
__device__ __forceinline__ v16bf load_a_frag(const float* __restrict__ p, int lane) {
  const int k0 = (lane & 16) ? 16 : 0;
  v16bf f;
#pragma unroll
  for (int j = 0; j < 8; ++j) f[j] = (__bf16)p[k0 + j];
#pragma unroll
  for (int j = 0; j < 8; ++j) f[8 + j] = (__bf16)p[k0 + 8 + j];
  return f;
}

// ==========================================================================================
// Blocked GEMM: C = act(A[MxK] @ W[KxN] + bias (+ C if accum))
// Block tile 64x64: 4 waves, each wave owns 16 rows x 64 cols (4 accumulators).
// B panel staged per 64-K chunk into LDS **K-transposed** (Bs[col][k], bf16) so each
// B fragment is 32 contiguous 32B-aligned bytes -> wide ds_load_b128, not u16 gathers.
// A fragment loaded once per k-step (global_load_b128 + cvt) and reused across 4 WMMAs.
// grid = (N/64, M/64), block = 128
// ==========================================================================================
__global__ __launch_bounds__(128) void k_gemm(const float* __restrict__ A,
                                              const float* __restrict__ W,
                                              const float* __restrict__ bias,
                                              float* __restrict__ C,
                                              int M, int K, int N, int act, int accum) {
  __shared__ __align__(32) __bf16 Bs[64][64];       // [col][k], 8 KB
  const int tid  = threadIdx.x;
  const int lane = tid & 31;
  const int wave = tid >> 5;
  const int colBase = blockIdx.x * 64;
  const int rowBase = blockIdx.y * 64 + wave * 16;
  const int row = rowBase + (lane & 15);
  const int r0  = rowBase + ((lane & 16) ? 8 : 0);

  v8f c[4];
  if (accum) {
#pragma unroll
    for (int nt = 0; nt < 4; ++nt) {
      const int col = colBase + nt * 16 + (lane & 15);
#pragma unroll
      for (int r = 0; r < 8; ++r) c[nt][r] = C[(long)(r0 + r) * N + col];
    }
  } else {
#pragma unroll
    for (int nt = 0; nt < 4; ++nt)
#pragma unroll
      for (int r = 0; r < 8; ++r) c[nt][r] = 0.0f;
  }

  const float* ap = A + (long)row * K;
  for (int kc = 0; kc < K; kc += 64) {
    __syncthreads();                                // protect previous chunk's LDS reads
    // stage B chunk transposed: thread packs two K-adjacent values for one column
    for (int i = tid; i < 64 * 32; i += 128) {
      const int cc = i & 63;                        // column (coalesced global read)
      const int kk = (i >> 6) << 1;                 // even k
      Bs[cc][kk]     = (__bf16)W[(long)(kc + kk) * N + colBase + cc];
      Bs[cc][kk + 1] = (__bf16)W[(long)(kc + kk + 1) * N + colBase + cc];
    }
    __syncthreads();
    if (kc + 64 < K) __builtin_prefetch(ap + kc + 64, 0, 0);   // global_prefetch_b8
#pragma unroll
    for (int ks = 0; ks < 64; ks += 32) {
      const v16bf a = load_a_frag(ap + kc + ks, lane);
      const int kb = ks + ((lane & 16) ? 16 : 0);   // B 32x16: lanes 0-15 K 0..15, 16-31 K 16..31
#pragma unroll
      for (int nt = 0; nt < 4; ++nt) {
        const int col = nt * 16 + (lane & 15);
        const v16bf b = *reinterpret_cast<const v16bf*>(&Bs[col][kb]);  // 2x ds_load_b128
        c[nt] = __builtin_amdgcn_wmma_f32_16x16x32_bf16(false, a, false, b, (short)0, c[nt], false, false);
      }
    }
  }

#pragma unroll
  for (int nt = 0; nt < 4; ++nt) {
    const int col = colBase + nt * 16 + (lane & 15);
    const float bb = bias ? bias[col] : 0.0f;
#pragma unroll
    for (int r = 0; r < 8; ++r) {
      float v = c[nt][r] + bb;
      if (act == 1) v = fmaxf(v, 0.0f);
      else if (act == 2) v = sigmoidf(v);
      C[(long)(r0 + r) * N + col] = v;
    }
  }
}

// ==========================================================================================
// GINE fused GEMM: lin_e = EA@W+b; msg = relu(h[src]+lin_e); aggr[dst] += msg
// Same 64x64 blocking + transposed LDS B panel; never materializes lin_e.
// grid = (N/64, E/64), block = 128
// ==========================================================================================
__global__ __launch_bounds__(128) void k_gemm_gine(const float* __restrict__ EA,
                                                   const float* __restrict__ W,
                                                   const float* __restrict__ bias,
                                                   const float* __restrict__ Hin, long hstride,
                                                   const int* __restrict__ src,
                                                   const int* __restrict__ dst,
                                                   float* __restrict__ aggr,
                                                   int E, int K, int N) {
  __shared__ __align__(32) __bf16 Bs[64][64];
  const int tid  = threadIdx.x;
  const int lane = tid & 31;
  const int wave = tid >> 5;
  const int colBase = blockIdx.x * 64;
  const int rowBase = blockIdx.y * 64 + wave * 16;
  const int row = rowBase + (lane & 15);
  const int r0  = rowBase + ((lane & 16) ? 8 : 0);

  v8f c[4];
#pragma unroll
  for (int nt = 0; nt < 4; ++nt)
#pragma unroll
    for (int r = 0; r < 8; ++r) c[nt][r] = 0.0f;

  const float* ap = EA + (long)row * K;
  for (int kc = 0; kc < K; kc += 64) {
    __syncthreads();
    for (int i = tid; i < 64 * 32; i += 128) {
      const int cc = i & 63;
      const int kk = (i >> 6) << 1;
      Bs[cc][kk]     = (__bf16)W[(long)(kc + kk) * N + colBase + cc];
      Bs[cc][kk + 1] = (__bf16)W[(long)(kc + kk + 1) * N + colBase + cc];
    }
    __syncthreads();
#pragma unroll
    for (int ks = 0; ks < 64; ks += 32) {
      const v16bf a = load_a_frag(ap + kc + ks, lane);
      const int kb = ks + ((lane & 16) ? 16 : 0);
#pragma unroll
      for (int nt = 0; nt < 4; ++nt) {
        const int col = nt * 16 + (lane & 15);
        const v16bf b = *reinterpret_cast<const v16bf*>(&Bs[col][kb]);
        c[nt] = __builtin_amdgcn_wmma_f32_16x16x32_bf16(false, a, false, b, (short)0, c[nt], false, false);
      }
    }
  }

#pragma unroll
  for (int r = 0; r < 8; ++r) {
    const int e = r0 + r;
    const int s = src[e];
    const long dbase = (long)dst[e] * 128;
    const long sbase = (long)s * hstride;
#pragma unroll
    for (int nt = 0; nt < 4; ++nt) {
      const int col = colBase + nt * 16 + (lane & 15);
      const float v   = c[nt][r] + bias[col];
      const float msg = fmaxf(Hin[sbase + col] + v, 0.0f);
      atomicAdd(&aggr[dbase + col], msg);           // global_atomic_add_f32
    }
  }
}

// ---------------- small-K affine + relu (atom/bond embedding) ----------------
__global__ void k_lin_small(const float* __restrict__ X, const float* __restrict__ W,
                            const float* __restrict__ b, float* __restrict__ Y,
                            int M, int K, int N) {
  long i = (long)blockIdx.x * blockDim.x + threadIdx.x;
  if (i >= (long)M * N) return;
  int m = (int)(i / N), n = (int)(i % N);
  float acc = b[n];
  for (int k = 0; k < K; ++k) acc += X[(long)m * K + k] * W[k * N + n];
  Y[i] = fmaxf(acc, 0.0f);
}

__global__ void k_zero(float* p, long n) {
  long i = (long)blockIdx.x * blockDim.x + threadIdx.x;
  if (i < n) p[i] = 0.0f;
}
__global__ void k_fillu(unsigned* p, unsigned v, long n) {
  long i = (long)blockIdx.x * blockDim.x + threadIdx.x;
  if (i < n) p[i] = v;
}

// t = (1+eps)*h + aggr
__global__ void k_gine_t(const float* __restrict__ Hin, long hstride,
                         const float* __restrict__ aggr, const float* __restrict__ eps,
                         float* __restrict__ T, int n) {
  long i = (long)blockIdx.x * blockDim.x + threadIdx.x;
  if (i >= (long)n * 128) return;
  int node = (int)(i >> 7), c = (int)(i & 127);
  T[i] = (1.0f + eps[0]) * Hin[(long)node * hstride + c] + aggr[i];
}

// out = LN(relu(x + addb)) * g + b + resid   (128 cols; one wave per row, shuffle reduce)
__global__ void k_ln128(const float* __restrict__ X, const float* __restrict__ addb,
                        const float* __restrict__ g, const float* __restrict__ b,
                        const float* __restrict__ resid, long rstride,
                        float* __restrict__ out, long ostride, int rows) {
  int wave = (int)(((long)blockIdx.x * blockDim.x + threadIdx.x) >> 5);
  int lane = threadIdx.x & 31;
  if (wave >= rows) return;
  const float* xr = X + (long)wave * 128;
  float v[4]; float s = 0.0f;
#pragma unroll
  for (int j = 0; j < 4; ++j) {
    int c = lane + 32 * j;
    float t = xr[c] + (addb ? addb[c] : 0.0f);
    t = fmaxf(t, 0.0f);
    v[j] = t; s += t;
  }
  for (int off = 16; off > 0; off >>= 1) s += __shfl_xor(s, off, 32);
  float mean = s * (1.0f / 128.0f);
  float var = 0.0f;
#pragma unroll
  for (int j = 0; j < 4; ++j) { float d = v[j] - mean; var += d * d; }
  for (int off = 16; off > 0; off >>= 1) var += __shfl_xor(var, off, 32);
  var *= (1.0f / 128.0f);
  float rs = rsqrtf(var + 1e-5f);
#pragma unroll
  for (int j = 0; j < 4; ++j) {
    int c = lane + 32 * j;
    out[(long)wave * ostride + c] = (v[j] - mean) * rs * g[c] + b[c] + resid[(long)wave * rstride + c];
  }
}

// GAT per-node attention scores: a_src[n,h], a_dst[n,h]
__global__ void k_att(const float* __restrict__ xl, const float* __restrict__ att_src,
                      const float* __restrict__ att_dst, float* __restrict__ asrc,
                      float* __restrict__ adst, int n) {
  long i = (long)blockIdx.x * blockDim.x + threadIdx.x;
  if (i >= (long)n * 4) return;
  int node = (int)(i >> 2), h = (int)(i & 3);
  const float* x = xl + (long)node * 128 + h * 32;
  float s1 = 0.0f, s2 = 0.0f;
  for (int d = 0; d < 32; ++d) { float xv = x[d]; s1 += xv * att_src[h * 32 + d]; s2 += xv * att_dst[h * 32 + d]; }
  asrc[i] = s1; adst[i] = s2;
}

__device__ __forceinline__ unsigned enc_f(float f) {
  unsigned u = __float_as_uint(f);
  return (u & 0x80000000u) ? ~u : (u | 0x80000000u);
}
__device__ __forceinline__ float dec_f(unsigned e) {
  unsigned u = (e & 0x80000000u) ? (e & 0x7FFFFFFFu) : ~e;
  return __uint_as_float(u);
}

// logits + segment max (edges + self loops)
__global__ void k_logit(const float* __restrict__ asrc, const float* __restrict__ adst,
                        const int* __restrict__ src, const int* __restrict__ dst,
                        float* __restrict__ logit, unsigned* __restrict__ nmax, int E, int en) {
  long i = (long)blockIdx.x * blockDim.x + threadIdx.x;
  if (i >= (long)en * 4) return;
  int e = (int)(i >> 2), h = (int)(i & 3);
  int s = (e < E) ? src[e] : (e - E);
  int d = (e < E) ? dst[e] : (e - E);
  float l = asrc[(long)s * 4 + h] + adst[(long)d * 4 + h];
  l = (l > 0.0f) ? l : 0.2f * l;                       // leaky_relu 0.2
  logit[i] = l;
  atomicMax(&nmax[(long)d * 4 + h], enc_f(l));
}

__global__ void k_expsum(const float* __restrict__ logit, const int* __restrict__ src,
                         const int* __restrict__ dst, const unsigned* __restrict__ nmax,
                         float* __restrict__ expv, float* __restrict__ nsum, int E, int en) {
  long i = (long)blockIdx.x * blockDim.x + threadIdx.x;
  if (i >= (long)en * 4) return;
  int e = (int)(i >> 2), h = (int)(i & 3);
  int d = (e < E) ? dst[e] : (e - E);
  float m = dec_f(nmax[(long)d * 4 + h]);
  float ev = __expf(logit[i] - m);
  expv[i] = ev;
  atomicAdd(&nsum[(long)d * 4 + h], ev);
}

__global__ void k_gataggr(const float* __restrict__ xl, const float* __restrict__ expv,
                          const float* __restrict__ nsum, const int* __restrict__ src,
                          const int* __restrict__ dst, float* __restrict__ gatv, int E, int en) {
  long i = (long)blockIdx.x * blockDim.x + threadIdx.x;
  if (i >= (long)en * 128) return;
  int e = (int)(i >> 7), c = (int)(i & 127), h = c >> 5;
  int s = (e < E) ? src[e] : (e - E);
  int d = (e < E) ? dst[e] : (e - E);
  float alpha = expv[(long)e * 4 + h] / nsum[(long)d * 4 + h];
  atomicAdd(&gatv[(long)d * 128 + c], alpha * xl[(long)s * 128 + c]);
}

// ---------------- Set2Set ----------------
__global__ void k_lstm(const float* __restrict__ gates, float* __restrict__ cs,
                       float* __restrict__ hs, int nb) {
  long i = (long)blockIdx.x * blockDim.x + threadIdx.x;
  if (i >= (long)nb * 512) return;
  int b = (int)(i >> 9), j = (int)(i & 511);
  const float* g = gates + (long)b * 2048;
  float ig = sigmoidf(g[j]), fg = sigmoidf(g[512 + j]);
  float gg = tanhf(g[1024 + j]), og = sigmoidf(g[1536 + j]);
  float c = fg * cs[i] + ig * gg;
  cs[i] = c;
  hs[i] = og * tanhf(c);
}

// e[n] = dot(xjk[n,:512], hs[batch[n],:])  -- one wave per node
__global__ void k_energy(const float* __restrict__ xjk, const float* __restrict__ hs,
                         const int* __restrict__ batch, float* __restrict__ e, int n) {
  int wave = (int)(((long)blockIdx.x * blockDim.x + threadIdx.x) >> 5);
  int lane = threadIdx.x & 31;
  if (wave >= n) return;
  int b = batch[wave];
  const float* x = xjk + (long)wave * 512;
  const float* h = hs + (long)b * 512;
  float s = 0.0f;
  for (int j = lane; j < 512; j += 32) s += x[j] * h[j];
  for (int off = 16; off > 0; off >>= 1) s += __shfl_xor(s, off, 32);
  if (lane == 0) e[wave] = s;
}

// each graph = exactly 32 contiguous nodes -> one wave per graph (perfect wave32 fit)
__global__ void k_gsoftmax(const float* __restrict__ e, float* __restrict__ a) {
  int b = blockIdx.x, lane = threadIdx.x;
  float v = e[(long)b * 32 + lane];
  float m = v;
  for (int off = 16; off > 0; off >>= 1) m = fmaxf(m, __shfl_xor(m, off, 32));
  float ev = __expf(v - m);
  float s = ev;
  for (int off = 16; off > 0; off >>= 1) s += __shfl_xor(s, off, 32);
  a[(long)b * 32 + lane] = ev / s;
}

__global__ void k_readout(const float* __restrict__ xjk, const float* __restrict__ a,
                          float* __restrict__ qstar) {
  int b = blockIdx.x;
  for (int f = threadIdx.x; f < 512; f += blockDim.x) {
    float s = 0.0f;
    for (int i = 0; i < 32; ++i) {
      int node = b * 32 + i;
      s += a[node] * xjk[(long)node * 512 + f];
    }
    qstar[(long)b * 1024 + 512 + f] = s;
  }
}
__global__ void k_qstar_h(const float* __restrict__ hs, float* __restrict__ qstar, int nb) {
  long i = (long)blockIdx.x * blockDim.x + threadIdx.x;
  if (i >= (long)nb * 512) return;
  int b = (int)(i >> 9), f = (int)(i & 511);
  qstar[(long)b * 1024 + f] = hs[i];
}

// LN over 256 cols (desc), one wave per row
__global__ void k_ln256(const float* __restrict__ X, const float* __restrict__ g,
                        const float* __restrict__ b, float* __restrict__ out, int rows) {
  int wave = (int)(((long)blockIdx.x * blockDim.x + threadIdx.x) >> 5);
  int lane = threadIdx.x & 31;
  if (wave >= rows) return;
  const float* x = X + (long)wave * 256;
  float v[8]; float s = 0.0f;
#pragma unroll
  for (int j = 0; j < 8; ++j) { v[j] = x[lane + 32 * j]; s += v[j]; }
  for (int off = 16; off > 0; off >>= 1) s += __shfl_xor(s, off, 32);
  float mean = s * (1.0f / 256.0f);
  float var = 0.0f;
#pragma unroll
  for (int j = 0; j < 8; ++j) { float d = v[j] - mean; var += d * d; }
  for (int off = 16; off > 0; off >>= 1) var += __shfl_xor(var, off, 32);
  var *= (1.0f / 256.0f);
  float rs = rsqrtf(var + 1e-5f);
#pragma unroll
  for (int j = 0; j < 8; ++j) {
    int c = lane + 32 * j;
    out[(long)wave * 256 + c] = (v[j] - mean) * rs * g[c] + b[c];
  }
}

// fused[b,:1024] = gamma*qstar+beta ; fused[b,1024:1152] = d2
__global__ void k_film(const float* __restrict__ film, const float* __restrict__ qstar,
                       const float* __restrict__ d2, float* __restrict__ fused, int nb) {
  long i = (long)blockIdx.x * blockDim.x + threadIdx.x;
  if (i >= (long)nb * 1152) return;
  int b = (int)(i / 1152), c = (int)(i % 1152);
  float v;
  if (c < 1024)
    v = film[(long)b * 2048 + c] * qstar[(long)b * 1024 + c] + film[(long)b * 2048 + 1024 + c];
  else
    v = d2[(long)b * 128 + (c - 1024)];
  fused[i] = v;
}

__global__ void k_zgate(float* __restrict__ z, const float* __restrict__ gate, long n) {
  long i = (long)blockIdx.x * blockDim.x + threadIdx.x;
  if (i < n) z[i] = fmaxf(z[i], 0.0f) * gate[i];
}

__global__ void k_head(const float* __restrict__ a2, const float* __restrict__ w,
                       const float* __restrict__ hb, float* __restrict__ out, int nb) {
  int i = blockIdx.x * blockDim.x + threadIdx.x;
  if (i >= nb) return;
  float s = hb[0];
  for (int k = 0; k < 256; ++k) s += a2[(long)i * 256 + k] * w[k];
  out[i] = s;
}

// ================================== host side ==================================
extern "C" void kernel_launch(void* const* d_in, const int* in_sizes, int n_in,
                              void* d_out, int out_size, void* d_ws, size_t ws_size,
                              hipStream_t stream) {
  (void)in_sizes; (void)n_in; (void)out_size; (void)ws_size;
  // ---- input pointers (JAX pytree flatten order: sorted dict keys at every level) ----
  const float* x         = (const float*)d_in[0];
  const float* edge_attr = (const float*)d_in[1];
  const float* desc      = (const float*)d_in[2];
  const float* ad1_b = (const float*)d_in[3],  * ad1_w = (const float*)d_in[4];
  const float* ad2_b = (const float*)d_in[5],  * ad2_w = (const float*)d_in[6];
  const float* atom_b = (const float*)d_in[7], * atom_w = (const float*)d_in[8];
  // blocks: 15 leaves each starting at 9:
  //   +0 att_dst +1 att_src +2 eps +3 gat_b +4 gat_w +5 ge_b +6 ge_w
  //   +7 m1_b +8 m1_w +9 m2_b +10 m2_w +11 ln1_b +12 ln1_g +13 ln2_b +14 ln2_g
  const float* bond_b = (const float*)d_in[69], * bond_w = (const float*)d_in[70];
  const float* d1_b = (const float*)d_in[71], * d1_w = (const float*)d_in[72];
  const float* d2_b = (const float*)d_in[73], * d2_w = (const float*)d_in[74];
  const float* dln_b = (const float*)d_in[75], * dln_g = (const float*)d_in[76];
  const float* film_b = (const float*)d_in[77], * film_w = (const float*)d_in[78];
  const float* f1_b = (const float*)d_in[79], * f1_w = (const float*)d_in[80];
  const float* f2_b = (const float*)d_in[81], * f2_w = (const float*)d_in[82];
  const float* g1_b = (const float*)d_in[83], * g1_w = (const float*)d_in[84];
  const float* g2_b = (const float*)d_in[85], * g2_w = (const float*)d_in[86];
  const float* hd_b = (const float*)d_in[87], * hd_w = (const float*)d_in[88];
  const float* b_hh = (const float*)d_in[89], * b_ih = (const float*)d_in[90];
  const float* w_hh = (const float*)d_in[91], * w_ih = (const float*)d_in[92];
  const int*   eidx  = (const int*)d_in[93];
  const int*   batch = (const int*)d_in[94];
  const int* srcp = eidx, * dstp = eidx + NEDGE;
  float* out = (float*)d_out;

  // ---- workspace layout ----
  float* ws = (float*)d_ws;
  size_t o = 0;
  auto alloc = [&](size_t nf) { float* p = ws + o; o += (nf + 63) & ~(size_t)63; return p; };
  float* H0    = alloc((size_t)NNODE * 128);
  float* XJK   = alloc((size_t)NNODE * 512);
  float* EA    = alloc((size_t)NEDGE * 128);
  float* AGGR  = alloc((size_t)NNODE * 128);   // also reused as t2
  float* T     = alloc((size_t)NNODE * 128);
  float* U     = alloc((size_t)NNODE * 128);   // mlp1 out / xl
  float* GATV  = alloc((size_t)NNODE * 128);
  float* ASRC  = alloc((size_t)NNODE * 4);
  float* ADST  = alloc((size_t)NNODE * 4);
  float* LOGIT = alloc((size_t)EN_TOT * 4);
  float* EXPV  = alloc((size_t)EN_TOT * 4);
  unsigned* NMAX = (unsigned*)alloc((size_t)NNODE * 4);
  float* NSUM  = alloc((size_t)NNODE * 4);
  float* QSTAR = alloc((size_t)NBATCH * 1024);
  float* HS    = alloc((size_t)NBATCH * 512);
  float* CS    = alloc((size_t)NBATCH * 512);
  float* GATES = alloc((size_t)NBATCH * 2048);
  float* EV    = alloc((size_t)NNODE);
  float* AA    = alloc((size_t)NNODE);
  float* DLN   = alloc((size_t)NBATCH * 256);
  float* D1    = alloc((size_t)NBATCH * 128);
  float* D2    = alloc((size_t)NBATCH * 128);
  float* FILMB = alloc((size_t)NBATCH * 2048);
  float* FUSEB = alloc((size_t)NBATCH * 1152);
  float* GT1   = alloc((size_t)NBATCH * 256);
  float* GATE  = alloc((size_t)NBATCH * 256);
  float* FT1   = alloc((size_t)NBATCH * 256);
  float* FOUT  = alloc((size_t)NBATCH * 256);
  float* A1    = alloc((size_t)NBATCH * 128);
  float* A2    = alloc((size_t)NBATCH * 256);

  auto cdiv = [](long a, long b) { return (a + b - 1) / b; };
  auto gemm = [&](const float* A, const float* W, const float* bias, float* C,
                  int M, int K, int N, int act, int accum) {
    dim3 g(N / 64, M / 64);
    k_gemm<<<g, dim3(128), 0, stream>>>(A, W, bias, C, M, K, N, act, accum);
  };
  auto zero = [&](float* p, long n) { k_zero<<<cdiv(n, 256), 256, 0, stream>>>(p, n); };

  // ---- embeddings ----
  k_lin_small<<<cdiv((long)NNODE * 128, 256), 256, 0, stream>>>(x, atom_w, atom_b, H0, NNODE, 6, 128);
  k_lin_small<<<cdiv((long)NEDGE * 128, 256), 256, 0, stream>>>(edge_attr, bond_w, bond_b, EA, NEDGE, 4, 128);

  // ---- 4 GINE+GAT blocks; layer output l lives in XJK[:, l*128:(l+1)*128] (JK concat for free) ----
  const float* hprev = H0;
  long hstride = 128;
  for (int blk = 0; blk < 4; ++blk) {
    const int base = 9 + blk * 15;
    const float* att_dst = (const float*)d_in[base + 0];
    const float* att_src = (const float*)d_in[base + 1];
    const float* eps     = (const float*)d_in[base + 2];
    const float* gat_b   = (const float*)d_in[base + 3];
    const float* gat_w   = (const float*)d_in[base + 4];
    const float* ge_b    = (const float*)d_in[base + 5];
    const float* ge_w    = (const float*)d_in[base + 6];
    const float* m1_b    = (const float*)d_in[base + 7];
    const float* m1_w    = (const float*)d_in[base + 8];
    const float* m2_b    = (const float*)d_in[base + 9];
    const float* m2_w    = (const float*)d_in[base + 10];
    const float* ln1_b   = (const float*)d_in[base + 11];
    const float* ln1_g   = (const float*)d_in[base + 12];
    const float* ln2_b   = (const float*)d_in[base + 13];
    const float* ln2_g   = (const float*)d_in[base + 14];

    // GINE: fused WMMA GEMM + relu(h[src]+lin_e) + scatter-add
    zero(AGGR, (long)NNODE * 128);
    k_gemm_gine<<<dim3(2, NEDGE / 64), dim3(128), 0, stream>>>(
        EA, ge_w, ge_b, hprev, hstride, srcp, dstp, AGGR, NEDGE, 128, 128);
    k_gine_t<<<cdiv((long)NNODE * 128, 256), 256, 0, stream>>>(hprev, hstride, AGGR, eps, T, NNODE);
    gemm(T, m1_w, m1_b, U, NNODE, 128, 128, 1, 0);         // relu(t @ mlp1 + b)
    gemm(U, m2_w, m2_b, AGGR, NNODE, 128, 128, 0, 0);      // t2 (reuse AGGR)
    k_ln128<<<cdiv(NNODE, 4), 128, 0, stream>>>(AGGR, nullptr, ln1_g, ln1_b,
                                                hprev, hstride, T, 128, NNODE);  // t = LN(relu(t2)) + h

    // GAT
    gemm(T, gat_w, nullptr, U, NNODE, 128, 128, 0, 0);     // xl
    k_att<<<cdiv((long)NNODE * 4, 256), 256, 0, stream>>>(U, att_src, att_dst, ASRC, ADST, NNODE);
    k_fillu<<<cdiv((long)NNODE * 4, 256), 256, 0, stream>>>(NMAX, 0x007FFFFFu, (long)NNODE * 4); // enc(-inf)
    zero(NSUM, (long)NNODE * 4);
    zero(GATV, (long)NNODE * 128);
    k_logit<<<cdiv((long)EN_TOT * 4, 256), 256, 0, stream>>>(ASRC, ADST, srcp, dstp, LOGIT, NMAX, NEDGE, EN_TOT);
    k_expsum<<<cdiv((long)EN_TOT * 4, 256), 256, 0, stream>>>(LOGIT, srcp, dstp, NMAX, EXPV, NSUM, NEDGE, EN_TOT);
    k_gataggr<<<cdiv((long)EN_TOT * 128, 256), 256, 0, stream>>>(U, EXPV, NSUM, srcp, dstp, GATV, NEDGE, EN_TOT);
    float* hout = XJK + (long)blk * 128;                   // stride 512 (JK slot)
    k_ln128<<<cdiv(NNODE, 4), 128, 0, stream>>>(GATV, gat_b, ln2_g, ln2_b, T, 128, hout, 512, NNODE);
    hprev = hout; hstride = 512;
  }

  // ---- Set2Set (3 steps) ----
  zero(QSTAR, (long)NBATCH * 1024);
  zero(HS, (long)NBATCH * 512);
  zero(CS, (long)NBATCH * 512);
  for (int it = 0; it < 3; ++it) {
    gemm(QSTAR, w_ih, b_ih, GATES, NBATCH, 1024, 2048, 0, 0);
    gemm(HS,    w_hh, b_hh, GATES, NBATCH, 512,  2048, 0, 1);   // accumulate
    k_lstm<<<cdiv((long)NBATCH * 512, 256), 256, 0, stream>>>(GATES, CS, HS, NBATCH);
    k_energy<<<cdiv(NNODE, 4), 128, 0, stream>>>(XJK, HS, batch, EV, NNODE);
    k_gsoftmax<<<NBATCH, 32, 0, stream>>>(EV, AA);
    k_readout<<<NBATCH, 256, 0, stream>>>(XJK, AA, QSTAR);
    k_qstar_h<<<cdiv((long)NBATCH * 512, 256), 256, 0, stream>>>(HS, QSTAR, NBATCH);
  }

  // ---- descriptor encoder + FiLM + gated fusion head ----
  k_ln256<<<cdiv(NBATCH, 4), 128, 0, stream>>>(desc, dln_g, dln_b, DLN, NBATCH);
  gemm(DLN, d1_w, d1_b, D1, NBATCH, 256, 128, 1, 0);
  gemm(D1,  d2_w, d2_b, D2, NBATCH, 128, 128, 1, 0);
  gemm(D2, film_w, film_b, FILMB, NBATCH, 128, 2048, 0, 0);
  k_film<<<cdiv((long)NBATCH * 1152, 256), 256, 0, stream>>>(FILMB, QSTAR, D2, FUSEB, NBATCH);
  gemm(FUSEB, g1_w, g1_b, GT1,  NBATCH, 1152, 256, 1, 0);
  gemm(GT1,   g2_w, g2_b, GATE, NBATCH, 256,  256, 2, 0);       // sigmoid
  gemm(FUSEB, f1_w, f1_b, FT1,  NBATCH, 1152, 256, 1, 0);
  gemm(FT1,   f2_w, f2_b, FOUT, NBATCH, 256,  256, 0, 0);
  k_zgate<<<cdiv((long)NBATCH * 256, 256), 256, 0, stream>>>(FOUT, GATE, (long)NBATCH * 256);
  gemm(FOUT, ad1_w, ad1_b, A1, NBATCH, 256, 128, 1, 0);
  gemm(A1,   ad2_w, ad2_b, A2, NBATCH, 128, 256, 1, 0);
  k_head<<<cdiv(NBATCH, 256), 256, 0, stream>>>(A2, hd_w, hd_b, out, NBATCH);
}
